// MatchingNetwork_55946243997910
// MI455X (gfx1250) — compile-verified, matrix operand order checked by reading, not run on hardware
//
#include <hip/hip_runtime.h>
#include <hip/hip_bf16.h>
#include <math.h>

typedef __attribute__((ext_vector_type(16))) _Float16 v16h;
typedef __attribute__((ext_vector_type(8)))  _Float16 v8h;
typedef __attribute__((ext_vector_type(8)))  float    v8f;

#define TPB    256
#define NWAVES 8

// ---- shared-memory element offsets (f16 elements) --------------------------
// two ping-pong activation buffers padded to 208 pixels (13 m-tiles) * 64 ch
#define ACT_ELEMS 13312               // 208*64
#define ACT_A     0
#define ACT_B     13312
#define ZPAD_OFF  26624               // 64-element zero halo region
#define SMEM_ELEMS 26688

// ---- workspace byte offsets ------------------------------------------------
#define WS_PACKED    0
#define PACKED_ELEMS 36864            // f16 elements per layer (18*4*32*16)
#define WS_SS        221184           // 4 layers * (64 scale + 64 shift) f32
#define WS_SEMB      223232           // 800*64 f32
#define WS_QEMB      428032           // 160*64 f32
#define WS_ACC       468992           // 32 f32

struct PrepParams {
    const float* k2;
    const float* k3;
    const float* k4;
    const float* b[4];
    const float* g[4];
    const float* be[4];
    const float* m[4];
    const float* v[4];
};

// ---------------------------------------------------------------------------
// Prep: fold conv-bias + inference BN into per-channel scale/shift, and pack
// k2/k3/k4 into the per-lane v_wmma_f32_16x16x32_f16 B-fragment order.
// Packed index: ((kc*4 + ntile)*32 + lane)*16 + slot
//   k = kc*32 + (slot/8)*16 + (lane/16)*8 + (slot&7)   (k = (dy*3+dx)*64 + cin)
//   n = ntile*16 + (lane&15)                            (output channel)
// ---------------------------------------------------------------------------
__global__ void prep_kernel(PrepParams p, _Float16* __restrict__ packed,
                            float* __restrict__ ss) {
    const int tid = threadIdx.x;
    if (blockIdx.x < 3) {
        const float* w = blockIdx.x == 0 ? p.k2 : (blockIdx.x == 1 ? p.k3 : p.k4);
        _Float16* dst = packed + (size_t)blockIdx.x * PACKED_ELEMS;
        for (int i = tid; i < PACKED_ELEMS; i += TPB) {
            const int slot  = i & 15;
            const int lane  = (i >> 4) & 31;
            const int ntile = (i >> 9) & 3;
            const int kc    = i >> 11;
            const int grp   = lane >> 4;
            const int k     = kc * 32 + (slot >> 3) * 16 + grp * 8 + (slot & 7);
            const int n     = ntile * 16 + (lane & 15);
            dst[i] = (_Float16)w[k * 64 + n];
        }
    } else {
        // 256 threads == 4 layers * 64 channels
        const int l = tid >> 6, c = tid & 63;
        const float sc = p.g[l][c] * rsqrtf(p.v[l][c] + 1e-3f);
        const float sh = sc * (p.b[l][c] - p.m[l][c]) + p.be[l][c];
        ss[l * 128 + c]      = sc;
        ss[l * 128 + 64 + c] = sh;
    }
}

// ---------------------------------------------------------------------------
// One Cin=64 conv block as implicit GEMM on v_wmma_f32_16x16x32_f16.
// A-fragments: unconditional ds_load_b128 from per-lane precomputed patch
// addresses (out-of-bounds patches aliased to a zeroed 64-elem LDS halo).
// Then fused scale/shift + ReLU, f16 store to padded LDS, 2x2 VALID max-pool.
// ---------------------------------------------------------------------------
template <int H, int W, int H2, int W2>
__device__ __forceinline__ void conv_wmma_layer(
    _Float16* __restrict__ smem,          // LDS base
    const int inOff, const int outOff,    // activation regions (ping-pong)
    const _Float16* __restrict__ packedW, // global
    const float* __restrict__ scale,
    const float* __restrict__ shift,
    float* __restrict__ embOut)           // non-null only for final layer
{
    constexpr int NPIX   = H * W;
    constexpr int MT     = (NPIX + 15) >> 4;
    constexpr int NTILES = MT * 4;

    const int tid  = threadIdx.x;
    const int wave = tid >> 5;
    const int lane = tid & 31;
    const int mrow = lane & 15;     // A-matrix row within tile
    const int grp  = lane >> 4;     // half-wave selects K sub-range
    const bool ghi = grp != 0;

    const v16h* pw = (const v16h*)packedW;

    for (int t = wave; t < NTILES; t += NWAVES) {
        const int mtile = t >> 2;
        const int ntile = t & 3;
        const int m = mtile * 16 + mrow;
        const bool mvalid = m < NPIX;
        const int y = m / W, x = m % W;

        // per-lane base address (f16 elements) for each of the 9 patches,
        // clamped to the zero halo when out of bounds
        int pa[9];
#pragma unroll
        for (int patch = 0; patch < 9; ++patch) {
            const int iy = y + patch / 3 - 1;
            const int ix = x + patch % 3 - 1;
            const bool inb = mvalid && iy >= 0 && iy < H && ix >= 0 && ix < W;
            pa[patch] = inb ? (inOff + (iy * W + ix) * 64) : ZPAD_OFF;
        }

        v8f acc = {};
#pragma unroll
        for (int kc = 0; kc < 18; ++kc) {
            v16h a;
#pragma unroll
            for (int h = 0; h < 2; ++h) {
                // k0/8 = 4*kc + 2*h + grp ; patch = k0/64 ; chan = k0%64
                const int J  = 4 * kc + 2 * h;
                const int aLo = pa[J >> 3]       + (J & 7) * 8;
                const int aHi = pa[(J + 1) >> 3] + ((J + 1) & 7) * 8;
                const int addr = ghi ? aHi : aLo;               // v_cndmask
                const v8h frag = *(const v8h*)(smem + addr);    // ds_load_b128
#pragma unroll
                for (int j = 0; j < 8; ++j) a[h * 8 + j] = frag[j];
            }
            const v16h bfrag = pw[(kc * 4 + ntile) * 32 + lane]; // 32B/lane
            acc = __builtin_amdgcn_wmma_f32_16x16x32_f16(
                false, a, false, bfrag, (short)0, acc, false, false);
        }

        // D layout: vgpr r, lanes 0-15 -> M=r,N=lane; lanes 16-31 -> M=8+r
        const int n = ntile * 16 + mrow;
        const float sc = scale[n], sh = shift[n];
#pragma unroll
        for (int r = 0; r < 8; ++r) {
            const int mm = mtile * 16 + grp * 8 + r;   // padded buffer: no guard
            float v = acc[r] * sc + sh;
            v = v > 0.f ? v : 0.f;
            smem[outOff + mm * 64 + n] = (_Float16)v;
        }
    }
    __syncthreads();

    // 2x2 VALID max-pool: outOff -> inOff (input acts no longer needed)
    for (int i = tid; i < H2 * W2 * 64; i += TPB) {
        const int c = i & 63, pp = i >> 6;
        const int py = pp / W2, px = pp % W2;
        const _Float16* co = smem + outOff;
        const float v00 = (float)co[((2 * py    ) * W + 2 * px    ) * 64 + c];
        const float v01 = (float)co[((2 * py    ) * W + 2 * px + 1) * 64 + c];
        const float v10 = (float)co[((2 * py + 1) * W + 2 * px    ) * 64 + c];
        const float v11 = (float)co[((2 * py + 1) * W + 2 * px + 1) * 64 + c];
        const float mv = fmaxf(fmaxf(v00, v01), fmaxf(v10, v11));
        if (embOut) embOut[c] = mv;
        else        smem[inOff + i] = (_Float16)mv;
    }
    __syncthreads();
}

// ---------------------------------------------------------------------------
// Embed: one block per image. Layer1 (Cin=1) fused conv+BN+ReLU+pool in VALU,
// layers 2..4 via WMMA implicit GEMM, final 64-d embedding to global ws.
// ---------------------------------------------------------------------------
__global__ __launch_bounds__(TPB)
void embed_kernel(const float* __restrict__ xs, const float* __restrict__ xq,
                  const float* __restrict__ k1,
                  const _Float16* __restrict__ packed,
                  const float* __restrict__ ss,
                  float* __restrict__ semb, float* __restrict__ qemb) {
    __shared__ __align__(16) _Float16 smem[SMEM_ELEMS];
    __shared__ float sImg[28 * 28];
    __shared__ float sW[576];

    const int tid = threadIdx.x;
    const int idx = blockIdx.x;
    const float* img = (idx < 800) ? xs + (size_t)idx * 784
                                   : xq + (size_t)(idx - 800) * 784;
    float* emb = (idx < 800) ? semb + (size_t)idx * 64
                             : qemb + (size_t)(idx - 800) * 64;

    // prefetch L2-resident packed weights (global_prefetch_b8)
    __builtin_prefetch(packed + (size_t)tid * 432, 0, 1);

    for (int i = tid; i < 784; i += TPB) sImg[i] = img[i];
    for (int i = tid; i < 576; i += TPB) sW[i] = k1[i];
    for (int i = tid; i < 64; i += TPB) smem[ZPAD_OFF + i] = (_Float16)0.f;
    __syncthreads();

    // --- layer 1: 28x28x1 -> conv3x3(64)+BN+ReLU -> pool -> 14x14x64 (f16) ---
    const float* sc1 = ss;
    const float* sh1 = ss + 64;
    for (int i = tid; i < 14 * 14 * 64; i += TPB) {
        const int c = i & 63, pp = i >> 6;
        const int py = pp / 14, px = pp % 14;
        const float sc = sc1[c], sh = sh1[c];
        float best = 0.f;  // post-ReLU values are >= 0
#pragma unroll
        for (int sy = 0; sy < 2; ++sy)
#pragma unroll
            for (int sx = 0; sx < 2; ++sx) {
                const int y = 2 * py + sy, x = 2 * px + sx;
                float acc = 0.f;
#pragma unroll
                for (int dy = 0; dy < 3; ++dy)
#pragma unroll
                    for (int dx = 0; dx < 3; ++dx) {
                        const int iy = y + dy - 1, ix = x + dx - 1;
                        if (iy >= 0 && iy < 28 && ix >= 0 && ix < 28)
                            acc += sImg[iy * 28 + ix] * sW[(dy * 3 + dx) * 64 + c];
                    }
                float v = acc * sc + sh;
                v = v > 0.f ? v : 0.f;
                best = fmaxf(best, v);
            }
        smem[ACT_A + i] = (_Float16)best;
    }
    __syncthreads();

    // --- layers 2..4: WMMA implicit GEMM ---
    conv_wmma_layer<14, 14, 7, 7>(smem, ACT_A, ACT_B, packed + 0 * PACKED_ELEMS,
                                  ss + 128, ss + 192, nullptr);
    conv_wmma_layer<7, 7, 3, 3>(smem, ACT_A, ACT_B, packed + 1 * PACKED_ELEMS,
                                ss + 256, ss + 320, nullptr);
    conv_wmma_layer<3, 3, 1, 1>(smem, ACT_A, ACT_B, packed + 2 * PACKED_ELEMS,
                                ss + 384, ss + 448, emb);
}

// ---------------------------------------------------------------------------
// Head: per-batch attention / softmax / CE; per-batch accuracy count to ws.
// ---------------------------------------------------------------------------
__global__ void head_kernel(const float* __restrict__ semb,
                            const float* __restrict__ qemb,
                            const int* __restrict__ ys,
                            const int* __restrict__ yq,
                            float* __restrict__ out,
                            float* __restrict__ accBuf) {
    const int b = blockIdx.x;
    const int lane = threadIdx.x;

    __shared__ float ssup[25 * 64];
    __shared__ float sq[5 * 64];
    __shared__ float sims[5][25];
    __shared__ float ceArr[5];
    __shared__ int   okArr[5];

    for (int i = lane; i < 25 * 64; i += 32) ssup[i] = semb[(size_t)b * 25 * 64 + i];
    for (int i = lane; i < 5 * 64; i += 32)  sq[i]   = qemb[(size_t)b * 5 * 64 + i];
    __syncthreads();

    if (lane < 25) {
        float mag2 = 0.f;
        for (int d = 0; d < 64; ++d) {
            const float v = ssup[lane * 64 + d];
            mag2 += v * v;
        }
        mag2 = mag2 < 1e-10f ? 1e-10f : mag2;
        const float inv = rsqrtf(mag2);
        for (int q = 0; q < 5; ++q) {
            float dot = 0.f;
            for (int d = 0; d < 64; ++d) dot += sq[q * 64 + d] * ssup[lane * 64 + d];
            sims[q][lane] = dot * inv;
        }
    }
    __syncthreads();

    if (lane < 5) {
        const int q = lane;
        float mx = sims[q][0];
        for (int s = 1; s < 25; ++s) mx = fmaxf(mx, sims[q][s]);
        float preds[5] = {0.f, 0.f, 0.f, 0.f, 0.f};
        float sum = 0.f;
        for (int s = 0; s < 25; ++s) {
            const float e = expf(sims[q][s] - mx);
            sum += e;
            preds[ys[b * 25 + s]] += e;
        }
        float tot = 0.f;
        for (int c = 0; c < 5; ++c) {
            preds[c] /= sum;   // softmax-weighted one-hot sum
            tot += preds[c];
        }
        int am = 0;
        float bv = preds[0];
        for (int c = 1; c < 5; ++c)
            if (preds[c] > bv) { bv = preds[c]; am = c; }
        const int t = yq[b * 5 + q];
        float p = preds[t] / tot;
        p = fminf(fmaxf(p, 1e-7f), 1.0f);
        ceArr[q] = -logf(p);
        okArr[q] = (am == t) ? 1 : 0;
    }
    __syncthreads();

    if (lane == 0) {
        float ce = 0.f;
        int ok = 0;
        for (int q = 0; q < 5; ++q) { ce += ceArr[q]; ok += okArr[q]; }
        out[b] = ce / 5.0f;
        accBuf[b] = (float)ok;
    }
}

__global__ void finalize_kernel(const float* __restrict__ accBuf,
                                float* __restrict__ out) {
    if (threadIdx.x == 0 && blockIdx.x == 0) {
        float s = 0.f;
        for (int b = 0; b < 32; ++b) s += accBuf[b];  // deterministic order
        out[32] = s / 160.0f;
    }
}

// ---------------------------------------------------------------------------
extern "C" void kernel_launch(void* const* d_in, const int* in_sizes, int n_in,
                              void* d_out, int out_size, void* d_ws, size_t ws_size,
                              hipStream_t stream) {
    const float* xs = (const float*)d_in[0];
    const int*   ys = (const int*)d_in[1];
    const float* xq = (const float*)d_in[2];
    const int*   yq = (const int*)d_in[3];
    const float* k1 = (const float*)d_in[4];

    PrepParams pp;
    pp.k2 = (const float*)d_in[10];
    pp.k3 = (const float*)d_in[16];
    pp.k4 = (const float*)d_in[22];
    for (int l = 0; l < 4; ++l) {
        pp.b[l]  = (const float*)d_in[4 + 6 * l + 1];
        pp.g[l]  = (const float*)d_in[4 + 6 * l + 2];
        pp.be[l] = (const float*)d_in[4 + 6 * l + 3];
        pp.m[l]  = (const float*)d_in[4 + 6 * l + 4];
        pp.v[l]  = (const float*)d_in[4 + 6 * l + 5];
    }

    char* ws = (char*)d_ws;
    _Float16* packed = (_Float16*)(ws + WS_PACKED);
    float* ss   = (float*)(ws + WS_SS);
    float* semb = (float*)(ws + WS_SEMB);
    float* qemb = (float*)(ws + WS_QEMB);
    float* accB = (float*)(ws + WS_ACC);
    float* out  = (float*)d_out;

    prep_kernel<<<4, TPB, 0, stream>>>(pp, packed, ss);
    embed_kernel<<<960, TPB, 0, stream>>>(xs, xq, k1, packed, ss, semb, qemb);
    head_kernel<<<32, 32, 0, stream>>>(semb, qemb, ys, yq, out, accB);
    finalize_kernel<<<1, 32, 0, stream>>>(accB, out);
}
